// Attention_25134148616754
// MI455X (gfx1250) — matrix operand order repaired and checked by reading, not compiled, not run on hardware
//
#include <hip/hip_runtime.h>
#include <hip/hip_bf16.h>
#include <cstdint>

// ---------------------------------------------------------------------------
// Problem constants (from reference)
// ---------------------------------------------------------------------------
#define HEADS    16
#define DIM_HEAD 64
#define QDIM     1024
#define INNER    1024
#define BB       4
#define TT       2048
#define MTOT     (BB * TT)          // 8192 rows
#define SCALE_F  0.125f             // 64^-0.5

typedef __bf16 bf16;
typedef __attribute__((ext_vector_type(16))) __bf16 v16bf;
typedef __attribute__((ext_vector_type(8)))  float  v8f;

// D = A(16x32 bf16) * B(32x16 bf16) + C(16x16 f32)
__device__ __forceinline__ v8f wmma_bf16(v16bf a, v16bf b, v8f c) {
  return __builtin_amdgcn_wmma_f32_16x16x32_bf16(
      /*neg_a=*/false, a, /*neg_b=*/false, b,
      /*c_mod=*/(short)0, c, /*reuse_a=*/false, /*reuse_b=*/false);
}

// A-matrix k order used here (contiguous, not AMD's table order): VGPR r of lane (half,m) holds
// K = k0, k0+1 with k0 = 2r + 16*half (matches B fills)
__device__ __forceinline__ int a_k0(int r, int half) {
  return 2 * r + 16 * half;
}

__device__ __forceinline__ v8f v8f_zero() {
  v8f z = {0.f, 0.f, 0.f, 0.f, 0.f, 0.f, 0.f, 0.f};
  return z;
}

// LDS byte offset of a __shared__ object: generic LDS-aperture addresses
// carry the LDS offset in bits [31:0] (ISA 10.2 aperture mapping).
__device__ __forceinline__ uint32_t lds_off(const void* p) {
  return (uint32_t)(uintptr_t)p;
}

// Async copy 16B from global to LDS (per-lane), tracked by ASYNCcnt.
// CDNA5 ISA: GLOBAL_LOAD_ASYNC_TO_LDS_B128, GV mode.
__device__ __forceinline__ void async_copy_b128(uint32_t lds_addr,
                                                const void* gaddr) {
  asm volatile("global_load_async_to_lds_b128 %0, %1, off"
               :
               : "v"(lds_addr), "v"(gaddr)
               : "memory");
}

__device__ __forceinline__ void wait_asynccnt0() {
  asm volatile("s_wait_asynccnt 0" ::: "memory");
}

// ---------------------------------------------------------------------------
// Kernel 1: QKV projection.  X[8192,1024]f32 @ W[1024,1024]f32 -> bf16
//   z=0: q[b][h][t][d]  (pre-scaled by SCALE)
//   z=1: k[b][h][t][d]
//   z=2: vT[b][h][d][t] (transposed so PV B-frag pairs are contiguous)
// Block tile 128x128, 8 waves in 4(M)x2(N), each wave 32x64 (2x4 WMMA tiles).
// f32->bf16 conversion must go through the VALU, so these stay as normal
// cooperative loads (compiler emits global_load_b64 + v_cvt_pk_bf16_f32).
// ---------------------------------------------------------------------------
__global__ __launch_bounds__(256) void qkv_proj_kernel(
    const float* __restrict__ X,
    const float* __restrict__ Wq, const float* __restrict__ Wk,
    const float* __restrict__ Wv,
    bf16* __restrict__ qo, bf16* __restrict__ ko, bf16* __restrict__ vo) {
  __shared__ bf16 As[128][32];   // [m][k]
  __shared__ bf16 Ws[128][32];   // transposed: [n][k] so B pairs are adjacent

  const int tid  = threadIdx.x;
  const int lane = tid & 31;
  const int wv   = tid >> 5;
  const int half = lane >> 4;
  const int lm   = lane & 15;
  const int n0   = blockIdx.x * 128;
  const int m0   = blockIdx.y * 128;
  const int z    = blockIdx.z;
  const float* W = (z == 0) ? Wq : (z == 1 ? Wk : Wv);
  const int wave_m = wv & 3;   // 0..3 : rows wave_m*32
  const int wave_n = wv >> 2;  // 0..1 : cols wave_n*64

  v8f acc[2][4];
#pragma unroll
  for (int sm = 0; sm < 2; ++sm)
#pragma unroll
    for (int sn = 0; sn < 4; ++sn) acc[sm][sn] = v8f_zero();

  for (int k0 = 0; k0 < QDIM; k0 += 32) {
    __syncthreads();  // protect LDS of previous iteration's readers
    // Cooperative load + f32->bf16 convert: A tile 128x32
#pragma unroll
    for (int i = tid; i < 2048; i += 256) {
      int r = i >> 4, c = (i & 15) * 2;
      float2 f = *(const float2*)(X + (size_t)(m0 + r) * QDIM + k0 + c);
      As[r][c] = (bf16)f.x;
      As[r][c + 1] = (bf16)f.y;
    }
    // W tile 32x128 loaded coalesced, stored transposed [n][k]
#pragma unroll
    for (int i = tid; i < 2048; i += 256) {
      int kk = i >> 6, nn = (i & 63) * 2;
      float2 f = *(const float2*)(W + (size_t)(k0 + kk) * INNER + n0 + nn);
      Ws[nn][kk] = (bf16)f.x;
      Ws[nn + 1][kk] = (bf16)f.y;
    }
    if (k0 + 32 < QDIM)
      __builtin_prefetch(X + (size_t)(m0 + (tid >> 1)) * QDIM + k0 + 32, 0, 3);
    __syncthreads();

    v16bf af[2], bfm[4];
#pragma unroll
    for (int sm = 0; sm < 2; ++sm) {
      int mrow = wave_m * 32 + sm * 16 + lm;
#pragma unroll
      for (int r = 0; r < 8; ++r) {
        int kk = a_k0(r, half);
        af[sm][2 * r] = As[mrow][kk];
        af[sm][2 * r + 1] = As[mrow][kk + 1];
      }
    }
#pragma unroll
    for (int sn = 0; sn < 4; ++sn) {
      int ncol = wave_n * 64 + sn * 16 + lm;
#pragma unroll
      for (int r = 0; r < 8; ++r) {
        int kk = 2 * r + 16 * half;   // B-matrix 32x16 K mapping
        bfm[sn][2 * r] = Ws[ncol][kk];
        bfm[sn][2 * r + 1] = Ws[ncol][kk + 1];
      }
    }
#pragma unroll
    for (int sm = 0; sm < 2; ++sm)
#pragma unroll
      for (int sn = 0; sn < 4; ++sn)
        acc[sm][sn] = wmma_bf16(af[sm], bfm[sn], acc[sm][sn]);
  }

  const float scale = (z == 0) ? SCALE_F : 1.0f;
  bf16* out = (z == 0) ? qo : (z == 1 ? ko : vo);
#pragma unroll
  for (int sm = 0; sm < 2; ++sm)
#pragma unroll
    for (int sn = 0; sn < 4; ++sn)
#pragma unroll
      for (int p = 0; p < 8; ++p) {
        int mg = m0 + wave_m * 32 + sm * 16 + p + 8 * half;  // C-layout row
        int cg = n0 + wave_n * 64 + sn * 16 + lm;            // C-layout col
        int b = mg >> 11, t = mg & (TT - 1);
        int h = cg >> 6, d = cg & 63;
        float val = acc[sm][sn][p] * scale;
        size_t idx;
        if (z == 2)
          idx = (((size_t)(b * HEADS + h)) * DIM_HEAD + d) * TT + t;  // vT
        else
          idx = (((size_t)(b * HEADS + h)) * TT + t) * DIM_HEAD + d;  // q,k
        out[idx] = (bf16)val;
      }
}

// ---------------------------------------------------------------------------
// Kernel 2: flash attention per (b,h).  Grid (T/128, H, B), 8 waves/block,
// each wave owns 16 query rows.  K/V blocks (32 keys) are staged ONCE per
// workgroup into LDS with GLOBAL_LOAD_ASYNC_TO_LDS_B128 (all 8 waves attend
// the same keys -> 8x traffic reduction vs per-wave loads), then fragments
// are built from contiguous 32-byte LDS reads.
// ---------------------------------------------------------------------------
__global__ __launch_bounds__(256) void attn_kernel(
    const bf16* __restrict__ q, const bf16* __restrict__ k,
    const bf16* __restrict__ vt, const int* __restrict__ amask,
    bf16* __restrict__ aout) {
  __shared__ bf16 Ks[32][64];     // K rows jb..jb+31          (4 KB)
  __shared__ bf16 Vs[64][32];     // vT slice [d][k]           (4 KB)
  __shared__ bf16 Ps[8][16][32];  // per-wave P transpose      (8 KB)

  const int tid  = threadIdx.x;
  const int lane = tid & 31;
  const int wv   = tid >> 5;
  const int half = lane >> 4;
  const int lm   = lane & 15;
  const int h = blockIdx.y, b = blockIdx.z;
  const int qr0 = blockIdx.x * 128 + wv * 16;
  const size_t base  = ((size_t)(b * HEADS + h)) * TT * DIM_HEAD;  // q,k
  const size_t vbase = ((size_t)(b * HEADS + h)) * DIM_HEAD * TT;  // vT

  // Cooperative-load index split: K tile = 32 rows x 8 16B-chunks,
  // V tile = 64 rows x 4 16B-chunks; 256 threads move one chunk each.
  const int krow = tid >> 3, kchunk = tid & 7;   // Ks: 32x(8 chunks of 8 bf16)
  const int vrow = tid >> 2, vchunk = tid & 3;   // Vs: 64x(4 chunks of 8 bf16)
  const uint32_t ks_lds = lds_off(&Ks[krow][kchunk * 8]);
  const uint32_t vs_lds = lds_off(&Vs[vrow][vchunk * 8]);

  // Q A-fragments for d=0..31 and d=32..63 (pairs contiguous in row-major Q)
  v16bf qf[2];
#pragma unroll
  for (int g = 0; g < 2; ++g)
#pragma unroll
    for (int r = 0; r < 8; ++r) {
      int kk = g * 32 + a_k0(r, half);
      const bf16* p = q + base + (size_t)(qr0 + lm) * DIM_HEAD + kk;
      qf[g][2 * r] = p[0];
      qf[g][2 * r + 1] = p[1];
    }

  float rm[8], rl[8];
  v8f o[4];
#pragma unroll
  for (int p = 0; p < 8; ++p) { rm[p] = -3.0e38f; rl[p] = 0.f; }
#pragma unroll
  for (int dt = 0; dt < 4; ++dt) o[dt] = v8f_zero();
  const float NEG = -3.0e38f;

  for (int jb = 0; jb < TT; jb += 32) {
    __syncthreads();  // previous block's LDS readers are done
    // Async-stage K block (32x64) and V block (64x32) into LDS.
    async_copy_b128(ks_lds, k + base + (size_t)(jb + krow) * DIM_HEAD +
                                kchunk * 8);
    async_copy_b128(vs_lds, vt + vbase + (size_t)vrow * TT + jb + vchunk * 8);
    wait_asynccnt0();
    __syncthreads();

    v8f s[2];
#pragma unroll
    for (int hs = 0; hs < 2; ++hs) {  // key half-blocks of 16
      s[hs] = v8f_zero();
#pragma unroll
      for (int g = 0; g < 2; ++g) {   // d halves of 32
        v16bf kf;                      // B = K^T from LDS (32B contiguous)
        int keyrow = hs * 16 + lm;
#pragma unroll
        for (int r = 0; r < 8; ++r) {
          int dd = g * 32 + 2 * r + 16 * half;
          kf[2 * r] = Ks[keyrow][dd];
          kf[2 * r + 1] = Ks[keyrow][dd + 1];
        }
        s[hs] = wmma_bf16(qf[g], kf, s[hs]);
      }
      // mask: this lane's acc column is key (jb + hs*16 + lm)
      int mv = amask[b * TT + jb + hs * 16 + lm];
      if (mv == 0) {
#pragma unroll
        for (int p = 0; p < 8; ++p) s[hs][p] = NEG;
      }
    }
    // online softmax: rows p+8*half live entirely inside one 16-lane half
    float cm[8], alpha[8], rs[8];
#pragma unroll
    for (int p = 0; p < 8; ++p) cm[p] = fmaxf(s[0][p], s[1][p]);
#pragma unroll
    for (int off = 1; off < 16; off <<= 1)
#pragma unroll
      for (int p = 0; p < 8; ++p)
        cm[p] = fmaxf(cm[p], __shfl_xor(cm[p], off, 16));
#pragma unroll
    for (int p = 0; p < 8; ++p) {
      float mn = fmaxf(rm[p], cm[p]);
      alpha[p] = __expf(rm[p] - mn);
      rm[p] = mn;
      float e0 = __expf(s[0][p] - mn);
      float e1 = __expf(s[1][p] - mn);
      s[0][p] = e0; s[1][p] = e1;
      rs[p] = e0 + e1;
    }
#pragma unroll
    for (int off = 1; off < 16; off <<= 1)
#pragma unroll
      for (int p = 0; p < 8; ++p) rs[p] += __shfl_xor(rs[p], off, 16);
#pragma unroll
    for (int p = 0; p < 8; ++p) rl[p] = rl[p] * alpha[p] + rs[p];
#pragma unroll
    for (int dt = 0; dt < 4; ++dt)
#pragma unroll
      for (int p = 0; p < 8; ++p) o[dt][p] *= alpha[p];

    // transpose P (C layout -> A layout) through per-wave LDS
#pragma unroll
    for (int p = 0; p < 8; ++p) {
      Ps[wv][p + 8 * half][lm]      = (bf16)s[0][p];
      Ps[wv][p + 8 * half][lm + 16] = (bf16)s[1][p];
    }
    v16bf pf;
#pragma unroll
    for (int r = 0; r < 8; ++r) {
      int kk = a_k0(r, half);
      pf[2 * r] = Ps[wv][lm][kk];
      pf[2 * r + 1] = Ps[wv][lm][kk + 1];
    }
    // O += P * V  (B-frag pairs contiguous: Vs is [d][k])
#pragma unroll
    for (int dt = 0; dt < 4; ++dt) {
      v16bf vf;
#pragma unroll
      for (int r = 0; r < 8; ++r) {
        int kk = 2 * r + 16 * half;
        vf[2 * r] = Vs[dt * 16 + lm][kk];
        vf[2 * r + 1] = Vs[dt * 16 + lm][kk + 1];
      }
      o[dt] = wmma_bf16(pf, vf, o[dt]);
    }
  }

  // normalize and store attn-out as bf16 [b*T + t][h*64 + d]
#pragma unroll
  for (int dt = 0; dt < 4; ++dt)
#pragma unroll
    for (int p = 0; p < 8; ++p) {
      int trow = qr0 + p + 8 * half;
      int cg = h * 64 + dt * 16 + lm;
      float val = o[dt][p] / rl[p];
      aout[(size_t)(b * TT + trow) * INNER + cg] = (bf16)val;
    }
}

// ---------------------------------------------------------------------------
// Kernel 3: output projection.  aout[8192,1024]bf16 @ Wo[1024,1024]f32 + bo
// -> f32 d_out.  A tile is already bf16 -> async copy to LDS; W tile needs
// f32->bf16 conversion so it stays on the VALU path.
// ---------------------------------------------------------------------------
__global__ __launch_bounds__(256) void out_proj_kernel(
    const bf16* __restrict__ A, const float* __restrict__ Wo,
    const float* __restrict__ bo, float* __restrict__ out) {
  __shared__ bf16 As[128][32];
  __shared__ bf16 Ws[128][32];  // [n][k]

  const int tid = threadIdx.x, lane = tid & 31, wv = tid >> 5;
  const int half = lane >> 4, lm = lane & 15;
  const int n0 = blockIdx.x * 128, m0 = blockIdx.y * 128;
  const int wave_m = wv & 3, wave_n = wv >> 2;

  // A tile: 128 rows x 4 chunks of 16B; 256 threads move 2 chunks each.
  const int ar0 = tid >> 2, ac0 = tid & 3;          // chunk tid
  const int ar1 = (tid + 256) >> 2, ac1 = (tid + 256) & 3;
  const uint32_t as_lds0 = lds_off(&As[ar0][ac0 * 8]);
  const uint32_t as_lds1 = lds_off(&As[ar1][ac1 * 8]);

  v8f acc[2][4];
#pragma unroll
  for (int sm = 0; sm < 2; ++sm)
#pragma unroll
    for (int sn = 0; sn < 4; ++sn) acc[sm][sn] = v8f_zero();

  for (int k0 = 0; k0 < INNER; k0 += 32) {
    __syncthreads();
    async_copy_b128(as_lds0, A + (size_t)(m0 + ar0) * INNER + k0 + ac0 * 8);
    async_copy_b128(as_lds1, A + (size_t)(m0 + ar1) * INNER + k0 + ac1 * 8);
#pragma unroll
    for (int i = tid; i < 2048; i += 256) {
      int kk = i >> 6, nn = (i & 63) * 2;
      float2 f = *(const float2*)(Wo + (size_t)(k0 + kk) * QDIM + n0 + nn);
      Ws[nn][kk] = (bf16)f.x;
      Ws[nn + 1][kk] = (bf16)f.y;
    }
    wait_asynccnt0();
    __syncthreads();

    v16bf af[2], bfm[4];
#pragma unroll
    for (int sm = 0; sm < 2; ++sm) {
      int mrow = wave_m * 32 + sm * 16 + lm;
#pragma unroll
      for (int r = 0; r < 8; ++r) {
        int kk = a_k0(r, half);
        af[sm][2 * r] = As[mrow][kk];
        af[sm][2 * r + 1] = As[mrow][kk + 1];
      }
    }
#pragma unroll
    for (int sn = 0; sn < 4; ++sn) {
      int ncol = wave_n * 64 + sn * 16 + lm;
#pragma unroll
      for (int r = 0; r < 8; ++r) {
        int kk = 2 * r + 16 * half;
        bfm[sn][2 * r] = Ws[ncol][kk];
        bfm[sn][2 * r + 1] = Ws[ncol][kk + 1];
      }
    }
#pragma unroll
    for (int sm = 0; sm < 2; ++sm)
#pragma unroll
      for (int sn = 0; sn < 4; ++sn)
        acc[sm][sn] = wmma_bf16(af[sm], bfm[sn], acc[sm][sn]);
  }

#pragma unroll
  for (int sm = 0; sm < 2; ++sm)
#pragma unroll
    for (int sn = 0; sn < 4; ++sn)
#pragma unroll
      for (int p = 0; p < 8; ++p) {
        int mg = m0 + wave_m * 32 + sm * 16 + p + 8 * half;
        int cg = n0 + wave_n * 64 + sn * 16 + lm;
        out[(size_t)mg * QDIM + cg] = acc[sm][sn][p] + bo[cg];
      }
}

// ---------------------------------------------------------------------------
// Launcher
// ---------------------------------------------------------------------------
extern "C" void kernel_launch(void* const* d_in, const int* in_sizes, int n_in,
                              void* d_out, int out_size, void* d_ws,
                              size_t ws_size, hipStream_t stream) {
  const float* X     = (const float*)d_in[0];   // hidden_states
  const int*   amask = (const int*)d_in[1];     // attention_mask
  const float* Wq    = (const float*)d_in[2];
  const float* Wk    = (const float*)d_in[3];
  const float* Wv    = (const float*)d_in[4];
  const float* Wo    = (const float*)d_in[5];
  const float* bo    = (const float*)d_in[6];
  float* out = (float*)d_out;

  const size_t QKV_ELEMS = (size_t)BB * HEADS * TT * DIM_HEAD;  // 8M each
  bf16* qws = (bf16*)d_ws;
  bf16* kws = qws + QKV_ELEMS;
  bf16* vws = kws + QKV_ELEMS;
  bf16* aws = vws + QKV_ELEMS;

  qkv_proj_kernel<<<dim3(INNER / 128, MTOT / 128, 3), 256, 0, stream>>>(
      X, Wq, Wk, Wv, qws, kws, vws);
  attn_kernel<<<dim3(TT / 128, HEADS, BB), 256, 0, stream>>>(
      qws, kws, vws, amask, aws);
  out_proj_kernel<<<dim3(QDIM / 128, MTOT / 128), 256, 0, stream>>>(
      aws, Wo, bo, out);
}
